// CausalSelfAttention_27822798143850
// MI455X (gfx1250) — matrix operand order repaired and checked by reading, not compiled, not run on hardware
//
#include <hip/hip_runtime.h>
#include <hip/hip_bf16.h>

typedef __attribute__((ext_vector_type(16))) _Float16 v16h;
typedef __attribute__((ext_vector_type(8)))  _Float16 v8h;
typedef __attribute__((ext_vector_type(8)))  float    v8f;
typedef __attribute__((ext_vector_type(4)))  unsigned v4u;
typedef __attribute__((ext_vector_type(8)))  int      v8i;
typedef __attribute__((ext_vector_type(4)))  int      v4i;

#define BATCH 4
#define SEQ   2048
#define DIM   1024
#define NH    16
#define HD    64
#define QKVLD (3 * DIM)   // 3072, row stride of qkv

#if __has_builtin(__builtin_amdgcn_tensor_load_to_lds) && __has_builtin(__builtin_amdgcn_s_wait_tensorcnt)
#define USE_TDM 1
#else
#define USE_TDM 0
#endif

static __device__ __forceinline__ void st_out(float* p, float v)    { *p = v; }
static __device__ __forceinline__ void st_out(_Float16* p, float v) { *p = (_Float16)v; }

static __device__ __forceinline__ v8f wmma_f16(const v16h& a, const v16h& b, const v8f& c) {
    return __builtin_amdgcn_wmma_f32_16x16x32_f16(false, a, false, b, (short)0, c, false, false);
}

static __device__ __forceinline__ v16h cat16(v8h lo, v8h hi) {
    return __builtin_shufflevector(lo, hi, 0, 1, 2, 3, 4, 5, 6, 7,
                                           8, 9, 10, 11, 12, 13, 14, 15);
}

// A-matrix fragment from a row-major [*,32] f16 LDS row: contiguous k {16g..16g+15}
static __device__ __forceinline__ v16h afrag_from_row(const _Float16* row, int g) {
    v8h lo = *(const v8h*)(row + 16 * g);
    v8h hi = *(const v8h*)(row + 16 * g + 8);
    return cat16(lo, hi);
}

// ---- DPP16 row_ror butterfly reductions over 16-lane rows (no LDS traffic) ----
template <int CTRL>
static __device__ __forceinline__ float dpp_f(float x) {
    return __int_as_float(
        __builtin_amdgcn_update_dpp(0, __float_as_int(x), CTRL, 0xf, 0xf, true));
}
static __device__ __forceinline__ float rowmax16(float x) {
    x = fmaxf(x, dpp_f<0x128>(x));   // row_ror:8
    x = fmaxf(x, dpp_f<0x124>(x));   // row_ror:4
    x = fmaxf(x, dpp_f<0x122>(x));   // row_ror:2
    x = fmaxf(x, dpp_f<0x121>(x));   // row_ror:1
    return x;
}
static __device__ __forceinline__ float rowsum16(float x) {
    x += dpp_f<0x128>(x);
    x += dpp_f<0x124>(x);
    x += dpp_f<0x122>(x);
    x += dpp_f<0x121>(x);
    return x;
}

// stage 8 contiguous elements with f32->f16 conversion (or straight f16 copy)
static __device__ __forceinline__ void stage8(const float* src, _Float16* dst) {
    const float4 a = *(const float4*)src;
    const float4 b = *(const float4*)(src + 4);
    v8h v = { (_Float16)a.x, (_Float16)a.y, (_Float16)a.z, (_Float16)a.w,
              (_Float16)b.x, (_Float16)b.y, (_Float16)b.z, (_Float16)b.w };
    *(v8h*)dst = v;
}
static __device__ __forceinline__ void stage8(const _Float16* src, _Float16* dst) {
    *(v8h*)dst = *(const v8h*)src;
}

#if USE_TDM
// Issue a TDM 2D tile load: tile_dim0 x tile_dim1 elements of 2 bytes,
// row stride `stride` elements, from global `gaddr` into LDS offset `lds_addr`.
static __device__ __forceinline__ void tdm_load_2d_f16(
    unsigned lds_addr, unsigned long long gaddr,
    unsigned tile_d0, unsigned tile_d1, unsigned stride_elems)
{
    v4u g0 = { 1u,                       // count=1, user descriptor
               lds_addr,                 // D#.lds_addr
               (unsigned)gaddr,          // global_addr[31:0]
               (unsigned)((gaddr >> 32) & 0x01FFFFFFu) | (2u << 30) };  // [56:32] | type=2
    v8i g1 = { (int)(1u << 16),                       // data_size=1 (2B), no mask/flags
               (int)((tile_d0 & 0xFFFFu) << 16),      // abar=0 | tensor_dim0.lo
               (int)((tile_d0 >> 16) | ((tile_d1 & 0xFFFFu) << 16)), // dim0.hi | dim1.lo
               (int)((tile_d1 >> 16) | (tile_d0 << 16)),             // dim1.hi | tile_dim0
               (int)tile_d1,                          // tile_dim1 | tile_dim2=0
               (int)stride_elems,                     // tensor_dim0_stride[31:0]
               0, 0 };                                // stride hi | dim1_stride
    v4i z4 = { 0, 0, 0, 0 };
#if __has_include(<hip/amd_detail/amd_gfx1250_TDM.h>)
    v8i z8 = { 0, 0, 0, 0, 0, 0, 0, 0 };
    __builtin_amdgcn_tensor_load_to_lds(g0, g1, z4, z4, z8, 0);   // clang-23 / therock arity
#else
    __builtin_amdgcn_tensor_load_to_lds(g0, g1, z4, z4, 0);       // ROCm 7.2 arity
#endif
}
#endif

// ---------------------------------------------------------------------------
// C[M,N] = A[M,K] * W[K,N] + bias[N]
// Block tile 64(M) x 128(N), 256 threads = 8 waves (4 M x 2 N subtiles),
// wave strip = 16x64 (4 f32 accumulators), K stepped by 32.
// A staged row-major [64][32]; W staged TRANSPOSED [128 cols][32 k] so B-fragment
// loads are contiguous 32B per lane (2x ds_load_b128, no scalar gathers).
// ---------------------------------------------------------------------------
template <typename TIn, typename TOut>
__global__ __launch_bounds__(256) void gemm_bias_wmma(
    const TIn* __restrict__ A, const float* __restrict__ W,
    const float* __restrict__ bias, TOut* __restrict__ C,
    int M, int N, int K)
{
    __shared__ alignas(64) _Float16 Asub[64 * 32];   // [row][k]
    __shared__ alignas(64) _Float16 Bt[128 * 32];    // [col][k]  (transposed)

    const int t      = threadIdx.x;
    const int lane   = t & 31;
    const int lane16 = lane & 15;
    const int g      = lane >> 4;
    const int w      = t >> 5;
    const int wm     = w & 3;
    const int wn     = w >> 2;
    const int blockN = blockIdx.x * 128;
    const int blockM = blockIdx.y * 64;

    // staging coordinates
    const int ar  = t >> 2;            // 0..63
    const int ac0 = (t & 3) * 8;       // 0/8/16/24
    const int br  = t >> 3;            // 0..31  (k within tile)
    const int bc0 = (t & 7) * 16;      // 0..112 (col chunk)

    v8f acc[4] = {};

    for (int kb = 0; kb < K; kb += 32) {
        // ---- stage A tile 64x32 (vector copy/convert) ----
        {
            const TIn* src = A + (size_t)(blockM + ar) * K + kb + ac0;
            stage8(src, &Asub[ar * 32 + ac0]);
            if (kb + 32 < K) __builtin_prefetch(src + 32, 0, 0);
        }
        // ---- stage W tile 32x128 transposed: vector global read, scattered LDS write ----
        {
            const float* src = W + (size_t)(kb + br) * N + blockN + bc0;
            if (kb + 32 < K) __builtin_prefetch(src + (size_t)32 * N, 0, 0);
            #pragma unroll
            for (int j = 0; j < 16; j += 4) {
                const float4 f = *(const float4*)(src + j);
                Bt[(bc0 + j + 0) * 32 + br] = (_Float16)f.x;
                Bt[(bc0 + j + 1) * 32 + br] = (_Float16)f.y;
                Bt[(bc0 + j + 2) * 32 + br] = (_Float16)f.z;
                Bt[(bc0 + j + 3) * 32 + br] = (_Float16)f.w;
            }
        }
        __syncthreads();

        const v16h af = afrag_from_row(&Asub[(wm * 16 + lane16) * 32], g);
        #pragma unroll
        for (int n = 0; n < 4; ++n) {
            const int col = wn * 64 + n * 16 + lane16;
            const v16h bf = *(const v16h*)(&Bt[col * 32 + g * 16]);  // contiguous per lane
            acc[n] = wmma_f16(af, bf, acc[n]);
        }
        __syncthreads();
    }

    // Epilogue: C layout — vgpr i => row i+8g, lane16 => col
    #pragma unroll
    for (int n = 0; n < 4; ++n) {
        const int col = blockN + wn * 64 + n * 16 + lane16;
        const float bv = bias[col];
        #pragma unroll
        for (int i = 0; i < 8; ++i) {
            const int row = blockM + wm * 16 + i + 8 * g;
            st_out(&C[(size_t)row * N + col], acc[n][i] + bv);
        }
    }
}

// ---------------------------------------------------------------------------
// Flash attention: one block per (b, h, 128-query strip); wave w owns 16 rows.
// K tile staged by the Tensor Data Mover (TDM) when available; V staged
// TRANSPOSED [d][key] manually. Waves skip key blocks entirely above their
// diagonal via a scalar (readfirstlane) branch — EXEC stays all-ones for WMMA.
// ---------------------------------------------------------------------------
__global__ __launch_bounds__(256) void attn_flash_wmma(
    const _Float16* __restrict__ qkv, _Float16* __restrict__ ctx)
{
    __shared__ alignas(64) _Float16 Ksub[32 * 64];     // [key][d]
    __shared__ alignas(64) _Float16 Vt[64 * 32];       // [d][key] (transposed)
    __shared__ alignas(64) _Float16 Pws[8][16 * 32];   // per-wave P transpose scratch

    const int t      = threadIdx.x;
    const int lane   = t & 31;
    const int lane16 = lane & 15;
    const int g      = lane >> 4;
    const int w      = t >> 5;

    const int qblk = blockIdx.x & 15;
    const int h    = (blockIdx.x >> 4) & 15;
    const int b    = blockIdx.x >> 8;
    const int qbase = qblk * 128;

    const _Float16* base = qkv + (size_t)b * SEQ * QKVLD;

    // Q fragments with fused scale: 0.125 * log2(e) -> logits already in log2 domain
    v16h qf[2];
    {
        const _Float16 qs = (_Float16)(0.125f * 1.44269504f);
        const _Float16* qrow = base + (size_t)(qbase + w * 16 + lane16) * QKVLD + h * HD;
        #pragma unroll
        for (int kk = 0; kk < 2; ++kk) {
            v8h lo = *(const v8h*)(qrow + kk * 32 + 16 * g);
            v8h hi = *(const v8h*)(qrow + kk * 32 + 16 * g + 8);
            qf[kk] = cat16(lo * qs, hi * qs);            // v_pk_mul_f16
        }
    }

    v8f acc[4] = {};
    float mrow[8], lrow[8];
    #pragma unroll
    for (int i = 0; i < 8; ++i) { mrow[i] = -__builtin_inff(); lrow[i] = 0.0f; }

    const int nkb = qbase / 32 + 4;   // cover keys up to qbase+127

    // staging coordinates
    const int kr = t >> 3;            // 0..31 (key)
    const int c0 = (t & 7) * 8;       // 0..56 (d chunk)

    for (int kb = 0; kb < nkb; ++kb) {
        const int kstart = kb * 32;
        // ---- stage K tile 32x64 via TDM (wave 0), V tile transposed (all waves) ----
#if USE_TDM
        if (__builtin_amdgcn_readfirstlane(w == 0 ? 1 : 0)) {
            const _Float16* kt = base + (size_t)kstart * QKVLD + DIM + h * HD;
            tdm_load_2d_f16((unsigned)(size_t)&Ksub[0],
                            (unsigned long long)(size_t)kt,
                            /*tile_d0=*/HD, /*tile_d1=*/32, /*stride=*/QKVLD);
            __builtin_amdgcn_s_wait_tensorcnt(0);
        }
#else
        {
            const _Float16* krow = base + (size_t)(kstart + kr) * QKVLD + DIM + h * HD + c0;
            stage8(krow, &Ksub[kr * 64 + c0]);
            if (kb + 1 < nkb) __builtin_prefetch(krow + (size_t)32 * QKVLD, 0, 0);
        }
#endif
        {
            const _Float16* vrow = base + (size_t)(kstart + kr) * QKVLD + 2 * DIM + h * HD + c0;
            const v8h vv = *(const v8h*)vrow;             // transposed scatter into Vt
            #pragma unroll
            for (int j = 0; j < 8; ++j)
                Vt[(c0 + j) * 32 + kr] = vv[j];
            if (kb + 1 < nkb) __builtin_prefetch(vrow + (size_t)32 * QKVLD, 0, 0);
        }
        __syncthreads();

        // Skip key blocks entirely above this wave's diagonal (uniform scalar branch)
        if (__builtin_amdgcn_readfirstlane(kstart <= qbase + w * 16 + 15 ? 1 : 0)) {
            // scores (log2 domain): 16q x 32keys = two 16x16 tiles, 2 WMMAs each
            v8f s0 = {}, s1 = {};
            #pragma unroll
            for (int kk = 0; kk < 2; ++kk) {
                const v16h bk0 = *(const v16h*)(&Ksub[lane16 * 64 + kk * 32 + g * 16]);
                const v16h bk1 = *(const v16h*)(&Ksub[(16 + lane16) * 64 + kk * 32 + g * 16]);
                s0 = wmma_f16(qf[kk], bk0, s0);
                s1 = wmma_f16(qf[kk], bk1, s1);
            }

            // causal mask + online softmax (DPP row reductions, exp2 domain)
            #pragma unroll
            for (int i = 0; i < 8; ++i) {
                const int rq = qbase + w * 16 + i + 8 * g;
                const int k0 = kstart + lane16;
                const int k1 = kstart + 16 + lane16;
                const float x0 = (k0 <= rq) ? s0[i] : -__builtin_inff();
                const float x1 = (k1 <= rq) ? s1[i] : -__builtin_inff();
                const float tm = rowmax16(fmaxf(x0, x1));
                const float mn = fmaxf(mrow[i], tm);
                const float p0 = __builtin_amdgcn_exp2f(x0 - mn);
                const float p1 = __builtin_amdgcn_exp2f(x1 - mn);
                const float rs = rowsum16(p0 + p1);
                const float alpha = __builtin_amdgcn_exp2f(mrow[i] - mn);
                lrow[i] = lrow[i] * alpha + rs;
                mrow[i] = mn;
                #pragma unroll
                for (int n = 0; n < 4; ++n) acc[n][i] *= alpha;
                Pws[w][(i + 8 * g) * 32 + lane16]      = (_Float16)p0;
                Pws[w][(i + 8 * g) * 32 + 16 + lane16] = (_Float16)p1;
            }

            // P: C-layout -> A-fragment (per-wave LDS region, DS in-order within wave)
            const v16h pf = afrag_from_row(&Pws[w][lane16 * 32], g);
            #pragma unroll
            for (int n = 0; n < 4; ++n) {
                const int col = n * 16 + lane16;
                const v16h vf = *(const v16h*)(&Vt[col * 32 + g * 16]);
                acc[n] = wmma_f16(pf, vf, acc[n]);
            }
        }
        __syncthreads();
    }

    // normalize + store ctx (f16) at [b, s, h*HD + d]
    #pragma unroll
    for (int i = 0; i < 8; ++i) {
        const float inv = 1.0f / lrow[i];
        const int rq = qbase + w * 16 + i + 8 * g;
        _Float16* crow = ctx + ((size_t)(b * SEQ + rq)) * DIM + h * HD;
        #pragma unroll
        for (int n = 0; n < 4; ++n)
            crow[n * 16 + lane16] = (_Float16)(acc[n][i] * inv);
    }
}

// ---------------------------------------------------------------------------
extern "C" void kernel_launch(void* const* d_in, const int* in_sizes, int n_in,
                              void* d_out, int out_size, void* d_ws, size_t ws_size,
                              hipStream_t stream) {
    const float* x      = (const float*)d_in[0];   // [B,S,D]
    const float* W_attn = (const float*)d_in[1];   // [D,3D]
    const float* b_attn = (const float*)d_in[2];   // [3D]
    const float* W_proj = (const float*)d_in[3];   // [D,D]
    const float* b_proj = (const float*)d_in[4];   // [D]
    float* out = (float*)d_out;

    _Float16* qkv = (_Float16*)d_ws;                                   // 48 MB
    _Float16* ctx = (_Float16*)((char*)d_ws +
                     (size_t)BATCH * SEQ * QKVLD * sizeof(_Float16));  // +16 MB

    const int M = BATCH * SEQ;   // 8192

    // 1) qkv = x @ W_attn + b_attn  (f16 out)
    gemm_bias_wmma<float, _Float16><<<dim3((3 * DIM) / 128, M / 64), dim3(256), 0, stream>>>(
        x, W_attn, b_attn, qkv, M, 3 * DIM, DIM);

    // 2) flash attention -> ctx (f16)
    attn_flash_wmma<<<dim3(BATCH * NH * (SEQ / 128)), dim3(256), 0, stream>>>(qkv, ctx);

    // 3) out = ctx @ W_proj + b_proj (f32 out)
    gemm_bias_wmma<_Float16, float><<<dim3(DIM / 128, M / 64), dim3(256), 0, stream>>>(
        ctx, W_proj, b_proj, out, M, DIM, DIM);
}
